// ColumnSelfAttention_10797547782881
// MI455X (gfx1250) — matrix operand order repaired and checked by reading, not compiled, not run on hardware
//
#include <hip/hip_runtime.h>
#include <hip/hip_bf16.h>

// ---------------------------------------------------------------------------
// Column self-attention for MI455X (gfx1250), bf16 WMMA pipeline.
//   x:[256,256,1,768] -> QKV proj -> per-(head,col) row attention -> O proj
// Outputs (concatenated in d_out): output f32 [256,256,1,768], probs f32
// [12,256,1,256,256].
// ---------------------------------------------------------------------------

#define T_TOK  65536   // R*C*B tokens
#define EMB    768
#define NQKV   2304
#define RR     256
#define CC     256
#define NH     12
#define HD     64
#define SCALING 0.125f // 64^-0.5

typedef __attribute__((ext_vector_type(16))) __bf16 v16bf;
typedef __attribute__((ext_vector_type(8)))  float  v8f;

union BFrag { unsigned u[8]; v16bf v; };
union U4U16 { uint4 v; unsigned short s[8]; };

__device__ __forceinline__ unsigned short f2bf(float f) {
    unsigned u = __float_as_uint(f);
    unsigned r = u + 0x7FFFu + ((u >> 16) & 1u);   // round-to-nearest-even
    return (unsigned short)(r >> 16);
}

// ---------------------------------------------------------------------------
// CDNA5 async global->LDS copy (ASYNCcnt path), guarded so the file still
// compiles if the toolchain lacks the builtin (falls back to uint4 copies).
// Probe round 2 told us the builtin's params are (int4 AS1*, int4 AS3*, Ii, Ii).
// ---------------------------------------------------------------------------
#if defined(__HIP_DEVICE_COMPILE__)
#if __has_builtin(__builtin_amdgcn_global_load_async_to_lds_b128) && \
    __has_builtin(__builtin_amdgcn_s_wait_asynccnt)
#define HAVE_ASYNC_LDS 1
#endif
#endif

typedef int v4i __attribute__((vector_size(16)));
typedef __attribute__((address_space(1))) v4i as1_v4i;
typedef __attribute__((address_space(3))) v4i as3_v4i;

__device__ __forceinline__ void async_copy16(const void* gsrc, void* ldst) {
#ifdef HAVE_ASYNC_LDS
    // generic->AS1 is a plain 64-bit reinterpret; generic shared pointer's
    // low 32 bits are the LDS offset (ISA flat-aperture rule) -> AS3.
    __builtin_amdgcn_global_load_async_to_lds_b128(
        (as1_v4i*)(unsigned long long)(uintptr_t)gsrc,
        (as3_v4i*)(unsigned)(uintptr_t)ldst, 0, 0);
#else
    *reinterpret_cast<uint4*>(ldst) = *reinterpret_cast<const uint4*>(gsrc);
#endif
}

__device__ __forceinline__ void async_wait() {
#ifdef HAVE_ASYNC_LDS
    __builtin_amdgcn_s_wait_asynccnt(0);
#endif
}

// ---------------------------------------------------------------------------
// Elementwise conversion / weight-packing kernels
// ---------------------------------------------------------------------------
__global__ void cast_f32_bf16(const float* __restrict__ in,
                              unsigned short* __restrict__ out, int n) {
    int i = blockIdx.x * blockDim.x + threadIdx.x;
    int stride = gridDim.x * blockDim.x;
    for (; i < n; i += stride) out[i] = f2bf(in[i]);
}

// wqkvT[k][n] (row-major [768 x 2304]); SCALING folded into the q slice.
__global__ void build_wqkvT(const float* __restrict__ Wq,
                            const float* __restrict__ Wk,
                            const float* __restrict__ Wv,
                            unsigned short* __restrict__ out) {
    int idx = blockIdx.x * blockDim.x + threadIdx.x;
    int stride = gridDim.x * blockDim.x;
    for (; idx < EMB * NQKV; idx += stride) {
        int k = idx / NQKV, n = idx % NQKV;
        float v;
        if (n < EMB)            v = Wq[n * EMB + k] * SCALING;
        else if (n < 2 * EMB)   v = Wk[(n - EMB) * EMB + k];
        else                    v = Wv[(n - 2 * EMB) * EMB + k];
        out[idx] = f2bf(v);
    }
}

__global__ void build_bqkv(const float* __restrict__ bq,
                           const float* __restrict__ bk,
                           const float* __restrict__ bv,
                           float* __restrict__ out) {
    int n = blockIdx.x * blockDim.x + threadIdx.x;
    if (n >= NQKV) return;
    if (n < EMB)            out[n] = bq[n] * SCALING;
    else if (n < 2 * EMB)   out[n] = bk[n - EMB];
    else                    out[n] = bv[n - 2 * EMB];
}

__global__ void build_woT(const float* __restrict__ Wo,
                          unsigned short* __restrict__ out) {
    int idx = blockIdx.x * blockDim.x + threadIdx.x;
    int stride = gridDim.x * blockDim.x;
    for (; idx < EMB * EMB; idx += stride) {
        int k = idx / EMB, n = idx % EMB;
        out[idx] = f2bf(Wo[n * EMB + k]);
    }
}

// ---------------------------------------------------------------------------
// Tiled bf16 GEMM: C[T x N] = A[T x K] * B[K x N] + bias.
// Block tile 128x64, 8 waves, per-wave 16x64 via 4 f32 accumulators.
// Double-buffered LDS: async A-tile copy + register-staged B slice for the
// next K-step overlap the WMMA stream; one barrier per iteration.
// ---------------------------------------------------------------------------
template <bool OUT_F32>
__global__ __launch_bounds__(256)
void gemm_bf16(const unsigned short* __restrict__ A,
               const unsigned short* __restrict__ B,
               const float* __restrict__ bias,
               void* __restrict__ Cout, int K, int N) {
    __shared__ unsigned short Atile[2][128][40];  // [buf][m][k], 20-bank stride
    __shared__ unsigned short BT[2][64][40];      // [buf][n][k] transposed

    const int tid  = threadIdx.x;
    const int lane = tid & 31;
    const int wave = tid >> 5;
    const int m0   = blockIdx.x * 128;
    const int n0   = blockIdx.y * 64;

    // loader thread mapping
    const int arow_ld = tid >> 2;          // 0..63 (two passes of 64 rows)
    const int acol_ld = (tid & 3) * 8;     // 0,8,16,24
    const int kk_ld   = tid >> 3;          // 0..31
    const int nn_ld   = (tid & 7) * 8;     // 0..56

    const v8f vzero = {0.f, 0.f, 0.f, 0.f, 0.f, 0.f, 0.f, 0.f};
    v8f acc[4] = {vzero, vzero, vzero, vzero};

    auto loadA_async = [&](int k0, int buf) {
#pragma unroll
        for (int p = 0; p < 2; ++p) {
            int r = p * 64 + arow_ld;
            async_copy16(A + (size_t)(m0 + r) * K + k0 + acol_ld,
                         &Atile[buf][r][acol_ld]);
        }
    };
    auto loadB_regs = [&](int k0) {
        return *reinterpret_cast<const uint4*>(
            B + (size_t)(k0 + kk_ld) * N + n0 + nn_ld);
    };
    auto storeB = [&](uint4 v, int buf) {
        U4U16 u; u.v = v;
#pragma unroll
        for (int e = 0; e < 8; ++e) BT[buf][nn_ld + e][kk_ld] = u.s[e];
    };

    // prologue: fill buffer 0
    loadA_async(0, 0);
    storeB(loadB_regs(0), 0);
    async_wait();
    __syncthreads();

    int buf = 0;
    for (int k0 = 0; k0 < K; k0 += 32) {
        const bool has_next = (k0 + 32 < K);
        uint4 bnext;
        if (has_next) {
            loadA_async(k0 + 32, buf ^ 1);   // async: no VGPR landing
            bnext = loadB_regs(k0 + 32);     // in-flight during compute
        }

        // --- A fragment (16x32, this wave's m-tile) ---
        BFrag a;
        int arow = wave * 16 + (lane & 15);
#pragma unroll
        for (int r = 0; r < 8; ++r) {
            int kc = ((lane >> 4) << 4) + (r << 1);
            a.u[r] = *reinterpret_cast<const unsigned*>(&Atile[buf][arow][kc]);
        }
        // --- 4 B fragments (32x16) + WMMA ---
#pragma unroll
        for (int nt = 0; nt < 4; ++nt) {
            BFrag b;
            int nn = nt * 16 + (lane & 15);
#pragma unroll
            for (int r = 0; r < 8; ++r) {
                int kk = ((lane >> 4) << 4) + (r << 1);
                b.u[r] = *reinterpret_cast<const unsigned*>(&BT[buf][nn][kk]);
            }
            acc[nt] = __builtin_amdgcn_wmma_f32_16x16x32_bf16(
                false, a.v, false, b.v, (short)0, acc[nt], false, false);
        }

        if (has_next) storeB(bnext, buf ^ 1);
        async_wait();
        __syncthreads();
        buf ^= 1;
    }

    // --- epilogue: bias + store (f32 C/D layout: row = r + 8*(lane/16)) ---
#pragma unroll
    for (int nt = 0; nt < 4; ++nt) {
        int col = n0 + nt * 16 + (lane & 15);
        float bv = bias ? bias[col] : 0.f;
#pragma unroll
        for (int r = 0; r < 8; ++r) {
            int row = m0 + wave * 16 + r + ((lane >> 4) << 3);
            float val = acc[nt][r] + bv;
            if (OUT_F32)
                reinterpret_cast<float*>(Cout)[(size_t)row * N + col] = val;
            else
                reinterpret_cast<unsigned short*>(Cout)[(size_t)row * N + col] = f2bf(val);
        }
    }
}

// ---------------------------------------------------------------------------
// Scores + softmax: one block per (h,c). S = q k^T (bf16 WMMA, f32 acc),
// mask, in-register softmax (shfl_xor over 16-lane halves), write f32 probs.
// k tile copied to LDS with async global->LDS (ASYNCcnt) ops.
// ---------------------------------------------------------------------------
__global__ __launch_bounds__(256)
void attn_scores(const unsigned short* __restrict__ qkv,
                 const unsigned char* __restrict__ mask,
                 float* __restrict__ probs) {
    __shared__ unsigned short kt[256][72];  // k[j][d], stride 72 u16 = 36 banks

    const int hc = blockIdx.x;
    const int h  = hc >> 8;
    const int c  = hc & 255;
    const int tid = threadIdx.x, lane = tid & 31, wave = tid >> 5;

    {   // load k rows: thread tid -> row j = tid (32 bf16 = 4 x 16B async)
        int j = tid;
        const unsigned short* src = qkv + (size_t)(j * CC + c) * NQKV + EMB + h * HD;
#pragma unroll
        for (int p = 0; p < 4; ++p)
            async_copy16(src + p * 8, &kt[j][p * 8]);
        async_wait();
    }
    __syncthreads();

    const size_t pbase = ((size_t)h * CC + c) * ((size_t)RR * RR);

    for (int mIter = 0; mIter < 2; ++mIter) {
        const int mbase = (wave * 2 + mIter) * 16;

        // q A-fragments (two K-steps over d=0..63) straight from global
        BFrag qf[2];
        {
            int qi = mbase + (lane & 15);
            const unsigned* q32 = reinterpret_cast<const unsigned*>(
                qkv + (size_t)(qi * CC + c) * NQKV + h * HD);
#pragma unroll
            for (int ks = 0; ks < 2; ++ks)
#pragma unroll
                for (int r = 0; r < 8; ++r) {
                    int d = ks * 32 + ((lane >> 4) << 4) + (r << 1);
                    qf[ks].u[r] = q32[d >> 1];
                }
        }

        const v8f vzero = {0.f, 0.f, 0.f, 0.f, 0.f, 0.f, 0.f, 0.f};
        v8f acc[16];
#pragma unroll
        for (int nt = 0; nt < 16; ++nt) acc[nt] = vzero;

#pragma unroll
        for (int nt = 0; nt < 16; ++nt) {
            int j = nt * 16 + (lane & 15);
#pragma unroll
            for (int ks = 0; ks < 2; ++ks) {
                BFrag b;  // k^T fragment: B[d][j] = k[j][d], pairs contiguous in d
#pragma unroll
                for (int r = 0; r < 8; ++r) {
                    int d = ks * 32 + ((lane >> 4) << 4) + (r << 1);
                    b.u[r] = *reinterpret_cast<const unsigned*>(&kt[j][d]);
                }
                acc[nt] = __builtin_amdgcn_wmma_f32_16x16x32_bf16(
                    false, qf[ks].v, false, b.v, (short)0, acc[nt], false, false);
            }
        }

        // padding mask (depends on (j,c) only)
#pragma unroll
        for (int nt = 0; nt < 16; ++nt) {
            int j = nt * 16 + (lane & 15);
            if (mask[(size_t)j * CC + c]) {
#pragma unroll
                for (int r = 0; r < 8; ++r) acc[nt][r] = -10000.f;
            }
        }

        // softmax: row M lives at vgpr r=M%8, lane-half M/8, lanes%16 = j%16
#pragma unroll
        for (int r = 0; r < 8; ++r) {
            float m = -3.4e38f;
#pragma unroll
            for (int nt = 0; nt < 16; ++nt) m = fmaxf(m, acc[nt][r]);
            for (int off = 1; off < 16; off <<= 1)
                m = fmaxf(m, __shfl_xor(m, off, 32));
            float s = 0.f;
#pragma unroll
            for (int nt = 0; nt < 16; ++nt) {
                float p = __expf(acc[nt][r] - m);
                acc[nt][r] = p;
                s += p;
            }
            for (int off = 1; off < 16; off <<= 1)
                s += __shfl_xor(s, off, 32);
            float inv = 1.0f / s;

            int i = mbase + r + ((lane >> 4) << 3);
            float* dst = probs + pbase + (size_t)i * RR;
#pragma unroll
            for (int nt = 0; nt < 16; ++nt)
                dst[nt * 16 + (lane & 15)] = acc[nt][r] * inv;
        }
    }
}

// ---------------------------------------------------------------------------
// Context: O = P * V per (h,c). P read back from f32 probs, cast to bf16
// fragments in registers; V staged TRANSPOSED in LDS (vT[d][j], 264-u16
// stride) so B-fragment K-pairs are contiguous u32 reads.
// ---------------------------------------------------------------------------
__global__ __launch_bounds__(256)
void attn_context(const unsigned short* __restrict__ qkv,
                  const float* __restrict__ probs,
                  unsigned short* __restrict__ ctx) {
    __shared__ unsigned short vT[64][264];  // vT[d][j]

    const int hc = blockIdx.x;
    const int h  = hc >> 8;
    const int c  = hc & 255;
    const int tid = threadIdx.x, lane = tid & 31, wave = tid >> 5;

    {   // thread tid owns v-row j = tid: read 4x uint4, scatter transposed
        int j = tid;
        const unsigned short* src = qkv + (size_t)(j * CC + c) * NQKV + 2 * EMB + h * HD;
#pragma unroll
        for (int p = 0; p < 4; ++p) {
            U4U16 u;
            u.v = *reinterpret_cast<const uint4*>(src + p * 8);
#pragma unroll
            for (int e = 0; e < 8; ++e) vT[p * 8 + e][j] = u.s[e];
        }
    }
    __syncthreads();

    const size_t pbase = ((size_t)h * CC + c) * ((size_t)RR * RR);

    for (int mIter = 0; mIter < 2; ++mIter) {
        const int mbase = (wave * 2 + mIter) * 16;
        const v8f vzero = {0.f, 0.f, 0.f, 0.f, 0.f, 0.f, 0.f, 0.f};
        v8f o[4] = {vzero, vzero, vzero, vzero};

        int i = mbase + (lane & 15);
        const float* prow = probs + pbase + (size_t)i * RR;

        for (int jt = 0; jt < 8; ++jt) {  // K = 256 over steps of 32 (j axis)
            BFrag p;  // P A-fragment, f32 -> bf16 on the fly
#pragma unroll
            for (int r = 0; r < 8; ++r) {
                int j0 = jt * 32 + ((lane >> 4) << 4) + (r << 1);
                float2 pv = *reinterpret_cast<const float2*>(prow + j0);
                p.u[r] = (unsigned)f2bf(pv.x) | ((unsigned)f2bf(pv.y) << 16);
            }
#pragma unroll
            for (int dt = 0; dt < 4; ++dt) {
                BFrag b;  // B[k=j][n=d] = vT[d][j]; K-pairs contiguous in j
                int d = dt * 16 + (lane & 15);
#pragma unroll
                for (int r = 0; r < 8; ++r) {
                    int j = jt * 32 + ((lane >> 4) << 4) + (r << 1);
                    b.u[r] = *reinterpret_cast<const unsigned*>(&vT[d][j]);
                }
                o[dt] = __builtin_amdgcn_wmma_f32_16x16x32_bf16(
                    false, p.v, false, b.v, (short)0, o[dt], false, false);
            }
        }

#pragma unroll
        for (int dt = 0; dt < 4; ++dt)
#pragma unroll
            for (int r = 0; r < 8; ++r) {
                int irow = mbase + r + ((lane >> 4) << 3);
                int d = dt * 16 + (lane & 15);
                ctx[(size_t)(irow * CC + c) * EMB + h * HD + d] = f2bf(o[dt][r]);
            }
    }
}

// ---------------------------------------------------------------------------
// Host launcher
// ---------------------------------------------------------------------------
extern "C" void kernel_launch(void* const* d_in, const int* in_sizes, int n_in,
                              void* d_out, int out_size, void* d_ws, size_t ws_size,
                              hipStream_t stream) {
    const float* x    = (const float*)d_in[0];
    const unsigned char* mask = (const unsigned char*)d_in[1];  // [B,R,C] bool
    const float* Wq = (const float*)d_in[2];
    const float* bq = (const float*)d_in[3];
    const float* Wk = (const float*)d_in[4];
    const float* bk = (const float*)d_in[5];
    const float* Wv = (const float*)d_in[6];
    const float* bv = (const float*)d_in[7];
    const float* Wo = (const float*)d_in[8];
    const float* bo = (const float*)d_in[9];

    float* out   = (float*)d_out;                       // [65536, 768]
    float* probs = out + (size_t)T_TOK * EMB;           // [12,256,1,256,256]

    // workspace carve-up (all chunks multiple of 256 B)
    char* ws = (char*)d_ws;
    unsigned short* xbf    = (unsigned short*)ws; ws += (size_t)T_TOK * EMB * 2;
    unsigned short* qkvbuf = (unsigned short*)ws; ws += (size_t)T_TOK * NQKV * 2;
    unsigned short* wqkvT  = (unsigned short*)ws; ws += (size_t)EMB * NQKV * 2;
    unsigned short* woT    = (unsigned short*)ws; ws += (size_t)EMB * EMB * 2;
    float*          bqkv   = (float*)ws;          ws += 16384;
    unsigned short* ctx    = (unsigned short*)ws;

    // 1) casts / weight packing
    cast_f32_bf16<<<4096, 256, 0, stream>>>(x, xbf, T_TOK * EMB);
    build_wqkvT<<<2048, 256, 0, stream>>>(Wq, Wk, Wv, wqkvT);
    build_bqkv<<<(NQKV + 255) / 256, 256, 0, stream>>>(bq, bk, bv, bqkv);
    build_woT<<<1024, 256, 0, stream>>>(Wo, woT);

    // 2) fused QKV projection: [65536,768] x [768,2304] -> bf16
    gemm_bf16<false><<<dim3(T_TOK / 128, NQKV / 64), 256, 0, stream>>>(
        xbf, wqkvT, bqkv, qkvbuf, EMB, NQKV);

    // 3) scores + softmax -> probs (f32, directly into d_out region)
    attn_scores<<<NH * CC, 256, 0, stream>>>(qkvbuf, mask, probs);

    // 4) context = P * V -> bf16
    attn_context<<<NH * CC, 256, 0, stream>>>(qkvbuf, probs, ctx);

    // 5) output projection: [65536,768] x [768,768] + bo -> f32 out
    gemm_bf16<true><<<dim3(T_TOK / 128, EMB / 64), 256, 0, stream>>>(
        ctx, woT, bo, out, EMB, EMB);
}
